// GraphEncoder_32624571581000
// MI455X (gfx1250) — matrix operand order repaired and checked by reading, not compiled, not run on hardware
//
#include <hip/hip_runtime.h>
#include <math.h>

// Gated graph encoder (N=128 nodes, F=32 feat, T=5 GRU steps) for MI455X/gfx1250.
// Memory-bound (0.5 FLOP/byte on per-edge 32x32 weight matrices). Strategy:
//  - compute only on real edges (~21% density)  -> ~5x traffic cut
//  - fuse the two U_u matvecs per edge (shared A tiles) -> 2x cut on U_u
//  - per-edge 32x32 matvec executed on the matrix engine:
//    2x v_wmma_f32_16x16x32_f16, B = vector broadcast across 16 columns
//  - deterministic reductions (fixed-order wave partials, no float atomics)
//  - double-buffered messages (global sync between timesteps via launches)

#define GN 128   // nodes
#define GF 32    // features (== wave32 lanes)
#define GT 5     // time steps
#define WAVES 8  // 256-thread workgroups

typedef __attribute__((ext_vector_type(16))) _Float16 v16h;
typedef __attribute__((ext_vector_type(8)))  float    v8f;

__device__ __forceinline__ float sigmoidf_(float x) {
  return 1.0f / (1.0f + __expf(-x));
}

// ---- WMMA 32x32 matvec helpers -------------------------------------------
// Vector form: "colvec" = lane l holds element l (F == 32 == wave32).
// A layout (16-bit 16x32, ISA 7.12.2): lane l<16 -> row l, K={0..7,16..23};
//   lane l>=16 -> row l-16, K={8..15,24..31}; 2 halves packed per VGPR, K asc.
// B layout (16-bit 32x16, per ISA striping): lanes 0-15 hold K={0..7,16..23} of
//   column N=lane; lanes 16-31 hold K={8..15,24..31}. All columns carry the same
//   vector, so D's columns are all equal to U_block @ v.
// D layout (32-bit 16x16): VGPR r: lanes 0-15 -> M=r, lanes 16-31 -> M=8+r.

__device__ __forceinline__ v16h build_B(const float* lbuf, int kb) {
  v16h B;
#pragma unroll
  for (int t = 0; t < 16; ++t) B[t] = (_Float16)lbuf[(kb >> 1) + (t & 7) + ((t >> 3) << 4)];
  return B;
}

__device__ __forceinline__ v16h build_A(const float* __restrict__ U, int r0, int wlane) {
  const int arow = r0 + (wlane & 15);
  const int cb = (wlane < 16) ? 0 : 8;
  const float* up = U + arow * GF + cb;
  v16h A;
#pragma unroll
  for (int t = 0; t < 8; ++t) A[t] = (_Float16)up[t];
#pragma unroll
  for (int t = 0; t < 8; ++t) A[8 + t] = (_Float16)up[16 + t];
  return A;
}

__device__ __forceinline__ void wmma_extract(const v8f& d, float* lbuf, int base, int wlane) {
  // All 16 columns equal; lane 0 holds rows base..base+7, lane 16 rows +8..+15.
  if (wlane == 0) {
#pragma unroll
    for (int t = 0; t < 8; ++t) lbuf[base + t] = d[t];
  } else if (wlane == 16) {
#pragma unroll
    for (int t = 0; t < 8; ++t) lbuf[base + 8 + t] = d[t];
  }
}

// out = U(32x32 row-major f32) @ v ; lbuf = per-wave 32-float LDS scratch.
// Same-wave LDS ops are in-order on CDNA5, so no barrier is needed.
__device__ __forceinline__ float matvec32(const float* __restrict__ U, float v,
                                          float* lbuf, int wlane) {
  lbuf[wlane] = v;
  const int kb = (wlane < 16) ? 0 : 16;
  v16h B = build_B(lbuf, kb);
  v8f c = {};
#pragma unroll
  for (int r0 = 0; r0 < GF; r0 += 16) {
    v16h A = build_A(U, r0, wlane);
    v8f d = __builtin_amdgcn_wmma_f32_16x16x32_f16(false, A, false, B,
                                                   (short)0, c, false, false);
    wmma_extract(d, lbuf, r0, wlane);
  }
  return lbuf[wlane];
}

// Two matvecs with the SAME matrix (shares A tiles -> halves U traffic).
// lbuf = per-wave 64-float LDS scratch.
__device__ __forceinline__ void matvec32x2(const float* __restrict__ U,
                                           float v1, float v2,
                                           float* lbuf, int wlane,
                                           float& o1, float& o2) {
  lbuf[wlane] = v1;
  lbuf[GF + wlane] = v2;
  const int kb = (wlane < 16) ? 0 : 16;
  v16h B1 = build_B(lbuf, kb);
  v16h B2 = build_B(lbuf + GF, kb);
  v8f c = {};
#pragma unroll
  for (int r0 = 0; r0 < GF; r0 += 16) {
    v16h A = build_A(U, r0, wlane);
    v8f d1 = __builtin_amdgcn_wmma_f32_16x16x32_f16(false, A, false, B1,
                                                    (short)0, c, false, false);
    v8f d2 = __builtin_amdgcn_wmma_f32_16x16x32_f16(false, A, false, B2,
                                                    (short)0, c, false, false);
    wmma_extract(d1, lbuf, r0, wlane);
    wmma_extract(d2, lbuf + GF, r0, wlane);
  }
  o1 = lbuf[wlane];
  o2 = lbuf[GF + wlane];
}

// ---- kernels --------------------------------------------------------------

__global__ void zero_kernel(float* p, size_t n) {
  size_t idx = (size_t)blockIdx.x * blockDim.x + threadIdx.x;
  size_t stride = (size_t)gridDim.x * blockDim.x;
  for (size_t k = idx; k < n; k += stride) p[k] = 0.0f;
}

// Hoisted node-feature terms: WuX[i,j] = W_u[i,j] @ x[i], WcmX likewise.
// One workgroup per row i; edges only (non-edge slots never read later).
__global__ void precompute_kernel(const float* __restrict__ x,
                                  const unsigned char* __restrict__ adj,
                                  const float* __restrict__ W_u,
                                  const float* __restrict__ W_cm,
                                  float* __restrict__ WuX,
                                  float* __restrict__ WcmX) {
  __shared__ float vscr[WAVES][2 * GF];
  __shared__ float xsh[GF];
  const int i = blockIdx.x;
  const int tid = threadIdx.x;
  const int w = tid >> 5, l = tid & 31;
  if (tid < GF) xsh[tid] = x[i * GF + tid];
  __syncthreads();
  const float xv = xsh[l];
  for (int j = w; j < GN; j += WAVES) {
    if (!adj[i * GN + j]) continue;  // uniform per wave -> EXEC stays full
    const size_t off = (size_t)(i * GN + j) * GF * GF;
    const size_t vo = (size_t)(i * GN + j) * GF + l;
    WuX[vo]  = matvec32(W_u  + off, xv, vscr[w], l);
    WcmX[vo] = matvec32(W_cm + off, xv, vscr[w], l);
  }
}

// One GRU timestep. Workgroup = row i (8 waves, 1 wave per edge at a time).
__global__ void gru_step_kernel(const float* __restrict__ msg_in,
                                float* __restrict__ msg_out,
                                const unsigned char* __restrict__ adj,
                                const float* __restrict__ U_u,
                                const float* __restrict__ U_cm,
                                const float* __restrict__ WuX,
                                const float* __restrict__ WcmX,
                                const float* __restrict__ b_u,
                                const float* __restrict__ b_cm) {
  __shared__ float s_in[GF];           // sum_k m[k,i]
  __shared__ float rowrm[GF];          // sum_k r[i,k]*m[i,k]
  __shared__ float part[WAVES][GF];    // deterministic reduction partials
  __shared__ float zbuf[GN * GF];      // z per edge slot        (16 KB)
  __shared__ float rmbuf[GN * GF];     // r*m per edge slot      (16 KB)
  __shared__ float vscr[WAVES][2 * GF];
  __shared__ int elist[GN];
  __shared__ int ecnt;

  const int i = blockIdx.x;
  const int tid = threadIdx.x;
  const int w = tid >> 5, l = tid & 31;

  // Deterministic edge list (serial scan by thread 0).
  if (tid == 0) {
    int c = 0;
    for (int k = 0; k < GN; ++k)
      if (adj[i * GN + k]) elist[c++] = k;
    ecnt = c;
  }

  // S_in[i] = sum_k msg_in[k,i] (messages are zero off-edge by construction).
  float acc = 0.0f;
  for (int k = w; k < GN; k += WAVES) acc += msg_in[((size_t)k * GN + i) * GF + l];
  part[w][l] = acc;
  __syncthreads();
  const int nE = ecnt;
  if (tid < GF) {
    float s = 0.0f;
    for (int ww = 0; ww < WAVES; ++ww) s += part[ww][tid];
    s_in[tid] = s;
  }
  __syncthreads();

  // Phase A: z, r, rm per edge; fused dual matvec shares U_u tiles.
  const float bu = b_u[l];
  float racc = 0.0f;
  for (int e = w; e < nE; e += WAVES) {
    const int j = elist[e];
    if (e + WAVES < nE)  // pull next edge's matrix while WMMAs run
      __builtin_prefetch(U_u + (size_t)(i * GN + elist[e + WAVES]) * GF * GF, 0, 1);
    const float* Uij = U_u + (size_t)(i * GN + j) * GF * GF;
    const float m  = msg_in[((size_t)i * GN + j) * GF + l];
    const float pv = s_in[l] - msg_in[((size_t)j * GN + i) * GF + l];
    float t1, t2;
    matvec32x2(Uij, pv, m, vscr[w], l, t1, t2);
    const float wux = WuX[((size_t)i * GN + j) * GF + l];
    const float z = sigmoidf_(wux + t1 + bu);
    const float r = sigmoidf_(wux + t2 + bu);
    const float rm = r * m;
    zbuf[e * GF + l] = z;
    rmbuf[e * GF + l] = rm;
    racc += rm;  // fixed order within a wave -> deterministic
  }
  part[w][l] = racc;
  __syncthreads();
  if (tid < GF) {
    float s = 0.0f;
    for (int ww = 0; ww < WAVES; ++ww) s += part[ww][tid];
    rowrm[tid] = s;
  }
  __syncthreads();

  // Phase B: current memory + GRU compose; write to ping-pong output buffer.
  const float bc = b_cm[l];
  for (int e = w; e < nE; e += WAVES) {
    const int j = elist[e];
    if (e + WAVES < nE)
      __builtin_prefetch(U_cm + (size_t)(i * GN + elist[e + WAVES]) * GF * GF, 0, 1);
    const float rs = rowrm[l] - rmbuf[e * GF + l];
    const float* Ucm = U_cm + (size_t)(i * GN + j) * GF * GF;
    const float t3 = matvec32(Ucm, rs, vscr[w], l);
    const float cm = tanhf(WcmX[((size_t)i * GN + j) * GF + l] + t3 + bc);
    const float pv = s_in[l] - msg_in[((size_t)j * GN + i) * GF + l];
    const float z = zbuf[e * GF + l];
    msg_out[((size_t)i * GN + j) * GF + l] = (1.0f - z) * pv + z * cm;
  }
}

// enc[i] = relu(U_nf[i]@x[i] + U_nm[i]@sum_j msg[i,j]); one wave per node.
__global__ void finalize_kernel(const float* __restrict__ msg,
                                const float* __restrict__ x,
                                const float* __restrict__ U_nf,
                                const float* __restrict__ U_nm,
                                float* __restrict__ out) {
  __shared__ float vscr[2 * GF];
  const int i = blockIdx.x;
  const int l = threadIdx.x;
  float s = 0.0f;
  for (int j = 0; j < GN; ++j) s += msg[((size_t)i * GN + j) * GF + l];
  const float xv = x[i * GF + l];
  const float a = matvec32(U_nf + (size_t)i * GF * GF, xv, vscr, l);
  const float b = matvec32(U_nm + (size_t)i * GF * GF, s, vscr, l);
  const float e = a + b;
  out[i * GF + l] = e > 0.0f ? e : 0.0f;
}

// ---- host glue ------------------------------------------------------------

extern "C" void kernel_launch(void* const* d_in, const int* in_sizes, int n_in,
                              void* d_out, int out_size, void* d_ws, size_t ws_size,
                              hipStream_t stream) {
  (void)in_sizes; (void)n_in; (void)out_size; (void)ws_size;
  const float* x            = (const float*)d_in[0];
  const unsigned char* adj  = (const unsigned char*)d_in[1];  // jax bool = 1 byte
  const float* W_u          = (const float*)d_in[2];
  const float* U_u          = (const float*)d_in[3];
  const float* W_cm         = (const float*)d_in[4];
  const float* U_cm         = (const float*)d_in[5];
  const float* b_u          = (const float*)d_in[6];
  const float* b_cm         = (const float*)d_in[7];
  const float* U_nf         = (const float*)d_in[8];
  const float* U_nm         = (const float*)d_in[9];
  float* out = (float*)d_out;

  float* wsf = (float*)d_ws;
  const size_t MSG = (size_t)GN * GN * GF;  // 2 MB each
  float* msgA = wsf;
  float* msgB = wsf + MSG;
  float* WuX  = wsf + 2 * MSG;
  float* WcmX = wsf + 3 * MSG;               // 8 MB total workspace

  // messages start at zero; non-edge slots stay zero forever.
  zero_kernel<<<2048, 256, 0, stream>>>(msgA, 2 * MSG);
  precompute_kernel<<<GN, WAVES * 32, 0, stream>>>(x, adj, W_u, W_cm, WuX, WcmX);

  const float* in_buf = msgA;
  float* out_buf = msgB;
  for (int t = 0; t < GT; ++t) {
    gru_step_kernel<<<GN, WAVES * 32, 0, stream>>>(in_buf, out_buf, adj,
                                                   U_u, U_cm, WuX, WcmX, b_u, b_cm);
    float* tmp = (float*)in_buf;
    in_buf = out_buf;
    out_buf = tmp;
  }
  finalize_kernel<<<GN, GF, 0, stream>>>(in_buf, x, U_nf, U_nm, out);
}
